// MolTransformerDual_85564338471040
// MI455X (gfx1250) — compile-verified, matrix operand order checked by reading, not run on hardware
//
#include <hip/hip_runtime.h>
#include <math.h>

// ---------------------------------------------------------------------------
// MolTransformerDual for MI455X (gfx1250, wave32, WMMA)
//
// Dense linears: v_wmma_f32_16x16x32_bf16, B-tiles staged in LDS via
// global_load_async_to_lds_b128 (ASYNCcnt) + bank-conflict-free padded layout.
// Scatter-softmax / segment-sum: global float atomics. LN/BN/L2: LDS trees.
//
// d_in layout (setup_inputs insertion order, params flattened in source
// order) — see round-0 comment block; indices used below are unchanged.
// d_out: [g_norm(2048x768) ; t_norm(2048x768)] f32
// ---------------------------------------------------------------------------

#define MTD_ASYNC_LDS 1   // flip off if assembler rejects async mnemonic

typedef __bf16 bf16_t;
typedef __attribute__((ext_vector_type(16))) __bf16 v16bf;
typedef __attribute__((ext_vector_type(8)))  float  v8f;

#define NN 100000
#define NE 250000
#define NG 2048

static __device__ __forceinline__ bf16_t f2bf(float f) {
  unsigned u = __float_as_uint(f);
  u += 0x7FFFu + ((u >> 16) & 1u);            // round-to-nearest-even
  unsigned short s = (unsigned short)(u >> 16);
  union { unsigned short u; bf16_t b; } cv; cv.u = s;
  return cv.b;
}

// K index inside a 16x32 bf16 A/B fragment for half h (0..15) of this lane.
static __device__ __forceinline__ int frag_k(int h, bool hi) {
  return hi ? (h < 8 ? 8 + h : 16 + h) : (h < 8 ? h : 8 + h);
}

static __device__ __forceinline__ v16bf load_frag_row(const bf16_t* p, bool hi) {
  v16bf f;
#pragma unroll
  for (int h = 0; h < 16; ++h) f[h] = p[frag_k(h, hi)];
  return f;
}

// ------------------------------ conversions --------------------------------
__global__ void mtd_f32_to_bf16(const float* __restrict__ src,
                                bf16_t* __restrict__ dst, int n) {
  int i = blockIdx.x * blockDim.x + threadIdx.x;
  if (i < n) dst[i] = f2bf(src[i]);
}

__global__ void mtd_zero_f32(float* __restrict__ p, int n) {
  int i = blockIdx.x * blockDim.x + threadIdx.x;
  if (i < n) p[i] = 0.0f;
}

__global__ void mtd_zero_u32(unsigned* __restrict__ p, int n) {
  int i = blockIdx.x * blockDim.x + threadIdx.x;
  if (i < n) p[i] = 0u;  // below every ordered-float key
}

// --------------------------- generic WMMA linear ---------------------------
// Y[M,Nout] = X[M,K](bf16) @ W[Nout,K](bf16)^T + bias ; f32 out.
// block = 256 thr (8 waves). wave -> 16 rows x 64 cols. grid.y = Nout/64.
// Weight tile (64 x KCHUNK) staged in LDS once per block per K-chunk via
// async-to-LDS loads; rows padded to 272B so b128 LDS reads are conflict-free.
#define KCHUNK 128
#define BPAD 136        // halves per LDS row (128 data + 8 pad)

__global__ void mtd_wmma_linear(const bf16_t* __restrict__ X,
                                const bf16_t* __restrict__ W,
                                const float* __restrict__ bias,
                                float* __restrict__ Y,
                                int M, int K, int Nout) {
  __shared__ __align__(16) bf16_t sh[64 * BPAD];
  const int tid  = threadIdx.x;
  const int lane = tid & 31;
  const int wave = tid >> 5;
  const bool hi  = lane >= 16;
  const int m0 = (blockIdx.x * 8 + wave) * 16;
  const int n0 = blockIdx.y * 64;
  const bool rows_valid = (m0 < M);

  v8f acc[4];
#pragma unroll
  for (int t = 0; t < 4; ++t)
#pragma unroll
    for (int j = 0; j < 8; ++j) acc[t][j] = 0.0f;

  int arow = m0 + (lane & 15);
  if (arow >= M) arow = M - 1;
  const bf16_t* xrowp = X + (size_t)arow * K;

  for (int kc0 = 0; kc0 < K; kc0 += KCHUNK) {
    // ---- cooperative stage of W[n0..n0+63, kc0..kc0+127] into LDS ----
#pragma unroll
    for (int s = 0; s < 4; ++s) {
      int seg = tid + s * 256;             // 1024 x 16B segments
      int row = seg >> 4;
      int c16 = seg & 15;
      const bf16_t* g = W + (size_t)(n0 + row) * K + kc0 + c16 * 8;
      bf16_t* l = &sh[row * BPAD + c16 * 8];
#if MTD_ASYNC_LDS
      unsigned lds_off = (unsigned)(uintptr_t)l;
      asm volatile("global_load_async_to_lds_b128 %0, %1, off"
                   :: "v"(lds_off), "v"((unsigned long long)(uintptr_t)g)
                   : "memory");
#else
      *(float4*)l = *(const float4*)g;
#endif
    }
#if MTD_ASYNC_LDS
    asm volatile("s_wait_asynccnt 0" ::: "memory");
#endif
    __syncthreads();

    // ---- 4 WMMA K-steps on the staged tile ----
    for (int kb = 0; kb < KCHUNK; kb += 32) {
      v16bf a = load_frag_row(xrowp + kc0 + kb, hi);
#pragma unroll
      for (int t = 0; t < 4; ++t) {
        int nl = t * 16 + (lane & 15);
        v16bf b = load_frag_row(sh + nl * BPAD + kb, hi);
        acc[t] = __builtin_amdgcn_wmma_f32_16x16x32_bf16(
            false, a, false, b, (short)0, acc[t], false, false);
      }
    }
    __syncthreads();
  }

  if (rows_valid) {
#pragma unroll
    for (int t = 0; t < 4; ++t) {
      int n = n0 + t * 16 + (lane & 15);
      float bv = bias ? bias[n] : 0.0f;
#pragma unroll
      for (int i = 0; i < 8; ++i) {
        int m = m0 + (hi ? 8 + i : i);
        if (m < M) Y[(size_t)m * Nout + n] = acc[t][i] + bv;
      }
    }
  }
}

// --------------------- fused embedding gather + projection -----------------
// Y[M,128] = concat_t(tab[t][idx[m,t]]) @ W[128,Ktot]^T + bias
struct TabPtrs { const bf16_t* t[9]; };

__global__ void mtd_embed_gemm(TabPtrs tabs, const int* __restrict__ idx,
                               int ncols, const bf16_t* __restrict__ W,
                               const float* __restrict__ bias,
                               float* __restrict__ Y, int M, int Ktot) {
  const int lane = threadIdx.x & 31;
  const int wave = threadIdx.x >> 5;
  const bool hi = lane >= 16;
  const int m0 = (blockIdx.x * 8 + wave) * 16;
  if (m0 >= M) return;

  v8f acc[8];
#pragma unroll
  for (int t = 0; t < 8; ++t)
#pragma unroll
    for (int j = 0; j < 8; ++j) acc[t][j] = 0.0f;

  int arow = m0 + (lane & 15);
  if (arow >= M) arow = M - 1;

  for (int kb = 0; kb < Ktot; kb += 32) {
    int tb = kb >> 7;          // table index (each table is 128 wide)
    int kc = kb & 127;         // column base within table
    int xi = idx[(size_t)arow * ncols + tb];
    v16bf a = load_frag_row(tabs.t[tb] + (size_t)xi * 128 + kc, hi);
#pragma unroll
    for (int c = 0; c < 8; ++c) {
      int n = c * 16 + (lane & 15);
      v16bf b = load_frag_row(W + (size_t)n * Ktot + kb, hi);
      acc[c] = __builtin_amdgcn_wmma_f32_16x16x32_bf16(
          false, a, false, b, (short)0, acc[c], false, false);
    }
  }
#pragma unroll
  for (int c = 0; c < 8; ++c) {
    int n = c * 16 + (lane & 15);
    float bv = bias[n];
#pragma unroll
    for (int i = 0; i < 8; ++i) {
      int m = m0 + (hi ? 8 + i : i);
      if (m < M) Y[(size_t)m * 128 + n] = acc[c][i] + bv;
    }
  }
}

// ------------------------------ attention ----------------------------------
__global__ void mtd_edge_alpha(const float4* __restrict__ q,
                               const float4* __restrict__ k,
                               const float4* __restrict__ el,
                               const int* __restrict__ src,
                               const int* __restrict__ dst,
                               float* __restrict__ alpha,
                               unsigned* __restrict__ amax, int E) {
  int i = blockIdx.x * blockDim.x + threadIdx.x;
  if (i >= E * 4) return;
  int e = i >> 2, h = i & 3;
  int s = src[e], d = dst[e];
  const float4* qp = q + (size_t)d * 32 + h * 8;
  const float4* kp = k + (size_t)s * 32 + h * 8;
  const float4* ep = el + (size_t)e * 32 + h * 8;
  float dot = 0.0f;
#pragma unroll
  for (int j = 0; j < 8; ++j) {
    float4 a = qp[j], b = kp[j], c = ep[j];
    dot += a.x * (b.x + c.x) + a.y * (b.y + c.y) +
           a.z * (b.z + c.z) + a.w * (b.w + c.w);
  }
  float al = dot * 0.1767766953f;  // 1/sqrt(32)
  alpha[i] = al;
  unsigned u = __float_as_uint(al);
  unsigned key = (u & 0x80000000u) ? ~u : (u | 0x80000000u);
  atomicMax(&amax[(size_t)d * 4 + h], key);
}

__global__ void mtd_edge_softmax_num(const float* __restrict__ alpha,
                                     const int* __restrict__ dst,
                                     const unsigned* __restrict__ amax,
                                     float* __restrict__ ex,
                                     float* __restrict__ den, int E) {
  int i = blockIdx.x * blockDim.x + threadIdx.x;
  if (i >= E * 4) return;
  int e = i >> 2, h = i & 3;
  int d = dst[e];
  unsigned key = amax[(size_t)d * 4 + h];
  unsigned u = (key & 0x80000000u) ? (key & 0x7FFFFFFFu) : ~key;
  float m = __uint_as_float(u);
  float v = __expf(alpha[i] - m);
  ex[i] = v;
  atomicAdd(&den[(size_t)d * 4 + h], v);
}

__global__ void mtd_edge_message(const float* __restrict__ v,
                                 const float* __restrict__ el,
                                 const float* __restrict__ ex,
                                 const float* __restrict__ den,
                                 const int* __restrict__ src,
                                 const int* __restrict__ dst,
                                 float* __restrict__ agg, int E) {
  int i = blockIdx.x * blockDim.x + threadIdx.x;
  if (i >= E * 128) return;
  int e = i >> 7, f = i & 127, h = f >> 5;
  int s = src[e], d = dst[e];
  float a = ex[(size_t)e * 4 + h] / (den[(size_t)d * 4 + h] + 1e-16f);
  float msg = (v[(size_t)s * 128 + f] + el[(size_t)e * 128 + f]) * a;
  atomicAdd(&agg[(size_t)d * 128 + f], msg);
}

// ------------- gated residual + layernorm + relu (per node) ----------------
__global__ void mtd_node_update(const float* __restrict__ agg,
                                const float* __restrict__ xr,
                                const float* __restrict__ hprev,
                                const float* __restrict__ betaW,
                                const float* __restrict__ ln_g,
                                const float* __restrict__ ln_b,
                                float* __restrict__ hout,
                                bf16_t* __restrict__ hout_bf, int N) {
  __shared__ float red[128];
  int node = blockIdx.x;
  int f = threadIdx.x;
  float o = agg[(size_t)node * 128 + f];
  float x = xr[(size_t)node * 128 + f];

  red[f] = o * betaW[f] + x * betaW[128 + f] + (o - x) * betaW[256 + f];
  __syncthreads();
  for (int s = 64; s > 0; s >>= 1) { if (f < s) red[f] += red[f + s]; __syncthreads(); }
  float beta = 1.0f / (1.0f + __expf(-red[0]));
  __syncthreads();

  float val = hprev[(size_t)node * 128 + f] + beta * x + (1.0f - beta) * o;

  red[f] = val;  __syncthreads();
  for (int s = 64; s > 0; s >>= 1) { if (f < s) red[f] += red[f + s]; __syncthreads(); }
  float mean = red[0] * (1.0f / 128.0f);
  __syncthreads();

  red[f] = val * val;  __syncthreads();
  for (int s = 64; s > 0; s >>= 1) { if (f < s) red[f] += red[f + s]; __syncthreads(); }
  float var = red[0] * (1.0f / 128.0f) - mean * mean;
  __syncthreads();

  float y = (val - mean) * rsqrtf(var + 1e-5f) * ln_g[f] + ln_b[f];
  y = fmaxf(y, 0.0f);
  hout[(size_t)node * 128 + f] = y;
  hout_bf[(size_t)node * 128 + f] = f2bf(y);
}

// ------------------------------- pooling -----------------------------------
__global__ void mtd_pool_feat(const float* __restrict__ h,
                              const int* __restrict__ batch,
                              float* __restrict__ s, int N) {
  int i = blockIdx.x * blockDim.x + threadIdx.x;
  if (i >= N * 128) return;
  int node = i >> 7, f = i & 127;
  atomicAdd(&s[(size_t)batch[node] * 128 + f], h[i]);
}

__global__ void mtd_pool_cnt(const int* __restrict__ batch,
                             float* __restrict__ cnt, int N) {
  int i = blockIdx.x * blockDim.x + threadIdx.x;
  if (i < N) atomicAdd(&cnt[batch[i]], 1.0f);
}

__global__ void mtd_pool_combine(const float* __restrict__ s,
                                 const float* __restrict__ cnt,
                                 float* __restrict__ g,
                                 bf16_t* __restrict__ gbf, int n) {
  int i = blockIdx.x * blockDim.x + threadIdx.x;
  if (i >= n) return;
  float c = fmaxf(cnt[i >> 7], 1.0f);
  float v = s[i] + s[i] / c;
  g[i] = v;
  gbf[i] = f2bf(v);
}

// ------------------------- batchnorm (eval stats) --------------------------
__global__ void mtd_colstats(const float* __restrict__ X, int M, int C,
                             float* __restrict__ mean, float* __restrict__ var) {
  __shared__ float rs[256], rs2[256];
  int c = blockIdx.x, t = threadIdx.x;
  float s = 0.0f, s2 = 0.0f;
  for (int r = t; r < M; r += 256) {
    float v = X[(size_t)r * C + c];
    s += v; s2 += v * v;
  }
  rs[t] = s; rs2[t] = s2; __syncthreads();
  for (int k = 128; k > 0; k >>= 1) {
    if (t < k) { rs[t] += rs[t + k]; rs2[t] += rs2[t + k]; }
    __syncthreads();
  }
  if (t == 0) {
    float m = rs[0] / (float)M;
    mean[c] = m;
    var[c] = rs2[0] / (float)M - m * m;
  }
}

__global__ void mtd_bn_relu(float* __restrict__ X,
                            const float* __restrict__ mean,
                            const float* __restrict__ var,
                            const float* __restrict__ g,
                            const float* __restrict__ b,
                            bf16_t* __restrict__ Xbf, int n, int C) {
  int i = blockIdx.x * blockDim.x + threadIdx.x;
  if (i >= n) return;
  int c = i % C;
  float y = (X[i] - mean[c]) * rsqrtf(var[c] + 1e-5f) * g[c] + b[c];
  y = fmaxf(y, 0.0f);
  X[i] = y;
  Xbf[i] = f2bf(y);
}

// ------------------------------ L2 normalize -------------------------------
__global__ void mtd_l2norm(const float* __restrict__ X, float* __restrict__ Y,
                           int C) {
  __shared__ float rs[256];
  int row = blockIdx.x, t = threadIdx.x;
  const float* xr = X + (size_t)row * C;
  float s = 0.0f;
  for (int c = t; c < C; c += 256) { float v = xr[c]; s += v * v; }
  rs[t] = s; __syncthreads();
  for (int k = 128; k > 0; k >>= 1) { if (t < k) rs[t] += rs[t + k]; __syncthreads(); }
  float inv = 1.0f / fmaxf(sqrtf(rs[0]), 1e-12f);
  for (int c = t; c < C; c += 256) Y[(size_t)row * C + c] = xr[c] * inv;
}

// ===========================================================================
extern "C" void kernel_launch(void* const* d_in, const int* in_sizes, int n_in,
                              void* d_out, int out_size, void* d_ws,
                              size_t ws_size, hipStream_t stream) {
  (void)in_sizes; (void)n_in; (void)out_size; (void)ws_size;

  static const int XCARD[9] = {119, 9, 11, 12, 9, 5, 8, 2, 2};
  static const int ECARD[3] = {22, 6, 2};

  const int* x          = (const int*)d_in[0];
  const int* e_src      = (const int*)d_in[1];
  const int* e_dst      = e_src + NE;
  const int* edge_attr  = (const int*)d_in[2];
  const int* batch      = (const int*)d_in[3];
  const float* text_emb = (const float*)d_in[4];

  size_t off = 0;
  auto alloc = [&](size_t bytes) -> char* {
    char* p = (char*)d_ws + off;
    off += (bytes + 255) & ~(size_t)255;
    return p;
  };

  bf16_t* w_nodep = (bf16_t*)alloc((size_t)128 * 1152 * 2);
  bf16_t* w_edgep = (bf16_t*)alloc((size_t)128 * 384 * 2);
  bf16_t* w_conv[3][5];
  for (int l = 0; l < 3; ++l)
    for (int j = 0; j < 5; ++j)
      w_conv[l][j] = (bf16_t*)alloc((size_t)128 * 128 * 2);
  bf16_t* w_gp1 = (bf16_t*)alloc((size_t)256 * 128 * 2);
  bf16_t* w_gp2 = (bf16_t*)alloc((size_t)768 * 256 * 2);
  bf16_t* w_tp1 = (bf16_t*)alloc((size_t)768 * 768 * 2);
  bf16_t* w_tp2 = (bf16_t*)alloc((size_t)768 * 768 * 2);
  bf16_t* ntab[9];
  for (int i = 0; i < 9; ++i) ntab[i] = (bf16_t*)alloc((size_t)XCARD[i] * 128 * 2);
  bf16_t* etab[3];
  for (int i = 0; i < 3; ++i) etab[i] = (bf16_t*)alloc((size_t)ECARD[i] * 128 * 2);

  float*  h     = (float*)alloc((size_t)NN * 128 * 4);
  bf16_t* h_bf  = (bf16_t*)alloc((size_t)NN * 128 * 2);
  float*  ebuf  = (float*)alloc((size_t)NE * 128 * 4);  // ea f32, reused as e-lin
  bf16_t* ea_bf = (bf16_t*)alloc((size_t)NE * 128 * 2);
  float*  qb    = (float*)alloc((size_t)NN * 128 * 4);
  float*  kb    = (float*)alloc((size_t)NN * 128 * 4);
  float*  vb    = (float*)alloc((size_t)NN * 128 * 4);
  float*  xrb   = (float*)alloc((size_t)NN * 128 * 4);
  float*  alpha = (float*)alloc((size_t)NE * 4 * 4);
  float*  exb   = (float*)alloc((size_t)NE * 4 * 4);
  unsigned* amax = (unsigned*)alloc((size_t)NN * 4 * 4);
  float*  den   = (float*)alloc((size_t)NN * 4 * 4);
  float*  agg   = (float*)alloc((size_t)NN * 128 * 4);
  float*  pools = (float*)alloc((size_t)NG * 128 * 4);
  float*  cnt   = (float*)alloc((size_t)NG * 4);
  float*  gfeat = (float*)alloc((size_t)NG * 128 * 4);
  bf16_t* gfeat_bf = (bf16_t*)alloc((size_t)NG * 128 * 2);
  float*  g1    = (float*)alloc((size_t)NG * 256 * 4);
  bf16_t* g1_bf = (bf16_t*)alloc((size_t)NG * 256 * 2);
  float*  gout  = (float*)alloc((size_t)NG * 768 * 4);
  bf16_t* t_bf  = (bf16_t*)alloc((size_t)NG * 768 * 2);
  float*  t1    = (float*)alloc((size_t)NG * 768 * 4);
  bf16_t* t1_bf = (bf16_t*)alloc((size_t)NG * 768 * 2);
  float*  tout  = (float*)alloc((size_t)NG * 768 * 4);
  float*  stat_m = (float*)alloc((size_t)768 * 4);
  float*  stat_v = (float*)alloc((size_t)768 * 4);

  auto cvt = [&](const void* src, bf16_t* dst, size_t n) {
    mtd_f32_to_bf16<<<dim3((unsigned)((n + 255) / 256)), dim3(256), 0, stream>>>(
        (const float*)src, dst, (int)n);
  };

  cvt(d_in[14], w_nodep, (size_t)128 * 1152);
  cvt(d_in[19], w_edgep, (size_t)128 * 384);
  for (int l = 0; l < 3; ++l) {
    int base = 21 + 13 * l;
    cvt(d_in[base + 0], w_conv[l][0], 128 * 128);  // q
    cvt(d_in[base + 2], w_conv[l][1], 128 * 128);  // k
    cvt(d_in[base + 4], w_conv[l][2], 128 * 128);  // v
    cvt(d_in[base + 6], w_conv[l][3], 128 * 128);  // e
    cvt(d_in[base + 8], w_conv[l][4], 128 * 128);  // skip
  }
  cvt(d_in[60], w_gp1, 256 * 128);
  cvt(d_in[64], w_gp2, 768 * 256);
  cvt(d_in[66], w_tp1, (size_t)768 * 768);
  cvt(d_in[70], w_tp2, (size_t)768 * 768);
  for (int i = 0; i < 9; ++i) cvt(d_in[5 + i], ntab[i], (size_t)XCARD[i] * 128);
  for (int i = 0; i < 3; ++i) cvt(d_in[16 + i], etab[i], (size_t)ECARD[i] * 128);

  TabPtrs ntp; for (int i = 0; i < 9; ++i) ntp.t[i] = ntab[i];
  TabPtrs etp; for (int i = 0; i < 3; ++i) etp.t[i] = etab[i];

  mtd_embed_gemm<<<dim3((NN + 127) / 128), dim3(256), 0, stream>>>(
      ntp, x, 9, w_nodep, (const float*)d_in[15], h, NN, 1152);
  cvt(h, h_bf, (size_t)NN * 128);

  mtd_embed_gemm<<<dim3((NE + 127) / 128), dim3(256), 0, stream>>>(
      etp, edge_attr, 3, w_edgep, (const float*)d_in[20], ebuf, NE, 384);
  cvt(ebuf, ea_bf, (size_t)NE * 128);

  dim3 gN((NN + 127) / 128, 2), gE((NE + 127) / 128, 2), blk(256);
  for (int l = 0; l < 3; ++l) {
    int base = 21 + 13 * l;
    mtd_wmma_linear<<<gN, blk, 0, stream>>>(h_bf, w_conv[l][0],
        (const float*)d_in[base + 1], qb, NN, 128, 128);
    mtd_wmma_linear<<<gN, blk, 0, stream>>>(h_bf, w_conv[l][1],
        (const float*)d_in[base + 3], kb, NN, 128, 128);
    mtd_wmma_linear<<<gN, blk, 0, stream>>>(h_bf, w_conv[l][2],
        (const float*)d_in[base + 5], vb, NN, 128, 128);
    mtd_wmma_linear<<<gN, blk, 0, stream>>>(h_bf, w_conv[l][4],
        (const float*)d_in[base + 9], xrb, NN, 128, 128);
    mtd_wmma_linear<<<gE, blk, 0, stream>>>(ea_bf, w_conv[l][3],
        (const float*)d_in[base + 7], ebuf, NE, 128, 128);

    mtd_zero_u32<<<dim3((NN * 4 + 255) / 256), blk, 0, stream>>>(amax, NN * 4);
    mtd_zero_f32<<<dim3((NN * 4 + 255) / 256), blk, 0, stream>>>(den, NN * 4);

    mtd_edge_alpha<<<dim3((NE * 4 + 255) / 256), blk, 0, stream>>>(
        (const float4*)qb, (const float4*)kb, (const float4*)ebuf,
        e_src, e_dst, alpha, amax, NE);
    mtd_edge_softmax_num<<<dim3((NE * 4 + 255) / 256), blk, 0, stream>>>(
        alpha, e_dst, amax, exb, den, NE);

    mtd_zero_f32<<<dim3((NN * 128 + 255) / 256), blk, 0, stream>>>(agg, NN * 128);
    mtd_edge_message<<<dim3((NE * 128 + 255) / 256), blk, 0, stream>>>(
        vb, ebuf, exb, den, e_src, e_dst, agg, NE);

    mtd_node_update<<<dim3(NN), dim3(128), 0, stream>>>(
        agg, xrb, h, (const float*)d_in[base + 10],
        (const float*)d_in[base + 11], (const float*)d_in[base + 12],
        h, h_bf, NN);
  }

  mtd_zero_f32<<<dim3((NG * 128 + 255) / 256), blk, 0, stream>>>(pools, NG * 128);
  mtd_zero_f32<<<dim3((NG + 255) / 256), blk, 0, stream>>>(cnt, NG);
  mtd_pool_feat<<<dim3((NN * 128 + 255) / 256), blk, 0, stream>>>(h, batch, pools, NN);
  mtd_pool_cnt<<<dim3((NN + 255) / 256), blk, 0, stream>>>(batch, cnt, NN);
  mtd_pool_combine<<<dim3((NG * 128 + 255) / 256), blk, 0, stream>>>(
      pools, cnt, gfeat, gfeat_bf, NG * 128);

  mtd_wmma_linear<<<dim3(NG / 128, 4), blk, 0, stream>>>(
      gfeat_bf, w_gp1, (const float*)d_in[61], g1, NG, 128, 256);
  mtd_colstats<<<dim3(256), blk, 0, stream>>>(g1, NG, 256, stat_m, stat_v);
  mtd_bn_relu<<<dim3((NG * 256 + 255) / 256), blk, 0, stream>>>(
      g1, stat_m, stat_v, (const float*)d_in[62], (const float*)d_in[63],
      g1_bf, NG * 256, 256);
  mtd_wmma_linear<<<dim3(NG / 128, 12), blk, 0, stream>>>(
      g1_bf, w_gp2, (const float*)d_in[65], gout, NG, 256, 768);

  cvt(text_emb, t_bf, (size_t)NG * 768);
  mtd_wmma_linear<<<dim3(NG / 128, 12), blk, 0, stream>>>(
      t_bf, w_tp1, (const float*)d_in[67], t1, NG, 768, 768);
  mtd_colstats<<<dim3(768), blk, 0, stream>>>(t1, NG, 768, stat_m, stat_v);
  mtd_bn_relu<<<dim3((NG * 768 + 255) / 256), blk, 0, stream>>>(
      t1, stat_m, stat_v, (const float*)d_in[68], (const float*)d_in[69],
      t1_bf, NG * 768, 768);
  mtd_wmma_linear<<<dim3(NG / 128, 12), blk, 0, stream>>>(
      t1_bf, w_tp2, (const float*)d_in[71], tout, NG, 768, 768);

  float* out_g = (float*)d_out;
  float* out_t = out_g + (size_t)NG * 768;
  mtd_l2norm<<<dim3(NG), blk, 0, stream>>>(gout, out_g, 768);
  mtd_l2norm<<<dim3(NG), blk, 0, stream>>>(tout, out_t, 768);
}
